// BiLstmCrfModel_79920751444231
// MI455X (gfx1250) — compile-verified, matrix operand order checked by reading, not run on hardware
//
#include <hip/hip_runtime.h>
#include <hip/hip_bf16.h>
#include <math.h>

// Problem constants (from reference)
constexpr int Vv = 30000;
constexpr int Ee = 64;      // embedding dim
constexpr int Hh = 64;      // hidden dim
constexpr int G4 = 256;     // 4*H gate width
constexpr int Kk = 9;       // tags
constexpr int Bb = 512;     // batch
constexpr int Tt = 512;     // time

typedef __attribute__((ext_vector_type(16))) _Float16 v16h;
typedef __attribute__((ext_vector_type(8)))  _Float16 v8h;
typedef __attribute__((ext_vector_type(8)))  float    v8f;

#define WMMA_F16(a, b, c) \
  __builtin_amdgcn_wmma_f32_16x16x32_f16(false, (a), false, (b), (short)0, (c), false, false)

constexpr float LOG2E = 1.4426950408889634f;

// Branch-free activations on v_exp_f32 / v_rcp_f32 (saturate correctly at +-inf).
__device__ __forceinline__ float fast_sigmoid(float z) {
  return __builtin_amdgcn_rcpf(1.f + __builtin_amdgcn_exp2f(-z * LOG2E));
}
__device__ __forceinline__ float fast_tanh(float z) {
  return 1.f - 2.f * __builtin_amdgcn_rcpf(1.f + __builtin_amdgcn_exp2f(2.f * z * LOG2E));
}

// Load a 16x32 f16 A-operand tile from a row-major [16][64] buffer (LDS or global).
// Per ISA layout: lane group g = lane/16 holds K in [8g,8g+8) and [16+8g,16+8g+8).
__device__ __forceinline__ v16h load_a_tile(const _Float16* buf, int lane, int kc) {
  const int row = lane & 15;
  const int g   = lane >> 4;
  const _Float16* p = buf + row * 64 + kc * 32 + g * 8;
  v8h lo = *(const v8h*)p;
  v8h hi = *(const v8h*)(p + 16);
  v16h r;
#pragma unroll
  for (int i = 0; i < 8; ++i) { r[i] = lo[i]; r[i + 8] = hi[i]; }
  return r;
}

// ---------------------------------------------------------------------------
// Fused bidirectional LSTM kernel.
// Grid: 64 workgroups of 256 threads. Blocks 0..31: forward over rows
// [16*blk, 16*blk+16); blocks 32..63: backward. Each WG iterates T steps,
// doing z = x@W + h@U + b as WMMA (8 wmma per wave per step), gates in f32
// VALU, h fed back via LDS in f16.
//
// Two-deep software pipeline: token for step s+1 is resident in a register
// (loaded at step s-1), so the emb row load for s+1 issues at the top of the
// WMMA phase with no dependent wait; token for s+2 is loaded with a full
// iteration of slack. Mask lives in registers (every thread holds its own
// row's token). 2 barriers per step.
// ---------------------------------------------------------------------------
__global__ __launch_bounds__(256) void bilstm_kernel(
    const int* __restrict__ tokens, const float* __restrict__ emb,
    const float* __restrict__ Wf, const float* __restrict__ Uf, const float* __restrict__ bf,
    const float* __restrict__ Wb, const float* __restrict__ Ub, const float* __restrict__ bb,
    _Float16* __restrict__ hf_out, _Float16* __restrict__ hb_out) {
  __shared__ float zbuf[16 * G4];                        // 16 KB gate pre-activations
  __shared__ alignas(16) _Float16 xbuf[2][16 * 64];     // double-buffered step input
  __shared__ alignas(16) _Float16 hbuf[16 * 64];        // h_{t-1} (f16)

  const int tid  = threadIdx.x;
  const int lane = tid & 31;
  const int wave = tid >> 5;
  const int g    = lane >> 4;
  const int nl   = lane & 15;
  const int dir  = blockIdx.x >> 5;          // 0 = fwd, 1 = bwd
  const int b0   = (blockIdx.x & 31) * 16;   // batch-row block

  const float* W    = dir ? Wb : Wf;
  const float* U    = dir ? Ub : Uf;
  const float* bias = dir ? bb : bf;
  _Float16*    hout = dir ? hb_out : hf_out;

  // --- Preload loop-invariant weight B-tiles into VGPRs (f32 -> f16) ------
  // Wave owns z columns [32*wave, 32*wave+32): two 16-col N-tiles, two K-chunks.
  v16h BW[2][2], BU[2][2];
#pragma unroll
  for (int nt = 0; nt < 2; ++nt) {
    const int col = wave * 32 + nt * 16 + nl;
#pragma unroll
    for (int kc = 0; kc < 2; ++kc) {
      v16h bw, bu;
#pragma unroll
      for (int i = 0; i < 8; ++i) {
        const int k1 = kc * 32 + g * 8 + i;
        const int k2 = k1 + 16;
        bw[i]     = (_Float16)W[k1 * G4 + col];
        bw[i + 8] = (_Float16)W[k2 * G4 + col];
        bu[i]     = (_Float16)U[k1 * G4 + col];
        bu[i + 8] = (_Float16)U[k2 * G4 + col];
      }
      BW[nt][kc] = bw;
      BU[nt][kc] = bu;
    }
  }

  // --- Per-thread gate element assignment: 4 contiguous (row, col) slots ---
  const int base = tid * 4;        // over 16*64 = 1024 h elements
  const int grow = base >> 6;      // 0..15 (batch row within block)
  const int gcol = base & 63;      // 0,4,...,60
  const int* tokrow = tokens + (b0 + grow) * Tt;  // this thread's token row
  float bI[4], bF[4], bG[4], bO[4];
#pragma unroll
  for (int i = 0; i < 4; ++i) {
    bI[i] = bias[gcol + i];
    bF[i] = bias[64 + gcol + i];
    bG[i] = bias[128 + gcol + i];
    bO[i] = bias[192 + gcol + i];
  }
  float creg[4] = {0.f, 0.f, 0.f, 0.f};
  float hreg[4] = {0.f, 0.f, 0.f, 0.f};

  // Prologue: token pipeline (cur = step 0, next = step 1), gather x0, zero h.
  int tok_cur  = tokrow[dir ? (Tt - 1) : 0];
  int tok_next = tokrow[dir ? (Tt - 2) : 1];
  {
    const float* er = emb + (size_t)tok_cur * Ee;
#pragma unroll
    for (int i = 0; i < 4; ++i) {
      xbuf[0][base + i] = (_Float16)er[gcol + i];
      hbuf[base + i]    = (_Float16)0.f;
    }
  }
  __syncthreads();

  for (int s = 0; s < Tt; ++s) {
    const int cur = s & 1;
    const int nxt = cur ^ 1;
    const int tt  = dir ? (Tt - 1 - s) : s;

    // A-operand tiles for this step.
    const v16h Ax0 = load_a_tile(xbuf[cur], lane, 0);
    const v16h Ax1 = load_a_tile(xbuf[cur], lane, 1);
    const v16h Ah0 = load_a_tile(hbuf, lane, 0);
    const v16h Ah1 = load_a_tile(hbuf, lane, 1);

    // Issue next-step emb row load immediately (token already resident), and
    // the token load for step s+2 (a full iteration of slack).
    float xr[4];
    if (s + 1 < Tt) {
      const float* er = emb + (size_t)tok_next * Ee;
#pragma unroll
      for (int i = 0; i < 4; ++i) xr[i] = er[gcol + i];
    }
    int tok_next2 = 0;
    if (s + 2 < Tt) tok_next2 = tokrow[dir ? (Tt - 3 - s) : (s + 2)];

    // z = x@W + h@U  (accumulated f32 via WMMA)
#pragma unroll
    for (int nt = 0; nt < 2; ++nt) {
      v8f acc = {};
      acc = WMMA_F16(Ax0, BW[nt][0], acc);
      acc = WMMA_F16(Ax1, BW[nt][1], acc);
      acc = WMMA_F16(Ah0, BU[nt][0], acc);
      acc = WMMA_F16(Ah1, BU[nt][1], acc);
      const int ncol = wave * 32 + nt * 16 + nl;
#pragma unroll
      for (int r = 0; r < 8; ++r) zbuf[(r + 8 * g) * G4 + ncol] = acc[r];
    }

    // Commit prefetched x into the other buffer.
    if (s + 1 < Tt) {
#pragma unroll
      for (int i = 0; i < 4; ++i) xbuf[nxt][base + i] = (_Float16)xr[i];
    }
    __syncthreads();

    // Gate nonlinearity + masked state update (branch-free f32 VALU).
    const bool mk = (tok_cur != 0);
#pragma unroll
    for (int i = 0; i < 4; ++i) {
      const float zi = zbuf[grow * G4 + gcol + i] + bI[i];
      const float zf = zbuf[grow * G4 + 64 + gcol + i] + bF[i];
      const float zg = zbuf[grow * G4 + 128 + gcol + i] + bG[i];
      const float zo = zbuf[grow * G4 + 192 + gcol + i] + bO[i];
      const float cn = fast_sigmoid(zf) * creg[i] + fast_sigmoid(zi) * fast_tanh(zg);
      const float hn = fast_sigmoid(zo) * fast_tanh(cn);
      creg[i] = mk ? cn : creg[i];
      hreg[i] = mk ? hn : hreg[i];
      const _Float16 hv = (_Float16)hreg[i];
      hbuf[base + i] = hv;  // safe: all hbuf A-reads completed before prior barrier
      hout[(size_t)((b0 + grow) * Tt + tt) * 64 + gcol + i] = hv;
    }
    tok_cur  = tok_next;
    tok_next = tok_next2;
    __syncthreads();
  }
}

// ---------------------------------------------------------------------------
// Potentials: pot[row, k] = [hf|hb][row, :128] @ Wd + bd, via WMMA with the
// N=16 tile zero-padded beyond k=9. Rows = B*T, K = 128 (4 chunks of 32).
// ---------------------------------------------------------------------------
__global__ __launch_bounds__(256) void potentials_kernel(
    const _Float16* __restrict__ hf, const _Float16* __restrict__ hb,
    const float* __restrict__ Wd, const float* __restrict__ bd,
    float* __restrict__ pot) {
  const int tid   = threadIdx.x;
  const int lane  = tid & 31;
  const int g     = lane >> 4;
  const int nl    = lane & 15;
  const int gwave = (blockIdx.x * 256 + tid) >> 5;
  const int nwaves = gridDim.x * 8;

  // Preload Wd B-tiles (zero-padded columns 9..15).
  v16h Bt[4];
#pragma unroll
  for (int kc = 0; kc < 4; ++kc) {
    v16h b;
#pragma unroll
    for (int i = 0; i < 8; ++i) {
      const int k1 = kc * 32 + g * 8 + i;
      const int k2 = k1 + 16;
      b[i]     = (nl < Kk) ? (_Float16)Wd[k1 * Kk + nl] : (_Float16)0.f;
      b[i + 8] = (nl < Kk) ? (_Float16)Wd[k2 * Kk + nl] : (_Float16)0.f;
    }
    Bt[kc] = b;
  }
  const float bias = (nl < Kk) ? bd[nl] : 0.f;

  const int ntiles = (Bb * Tt) / 16;
  for (int tile = gwave; tile < ntiles; tile += nwaves) {
    const _Float16* ra = hf + (size_t)tile * 16 * 64;
    const _Float16* rb = hb + (size_t)tile * 16 * 64;
    v8f acc = {};
    acc = WMMA_F16(load_a_tile(ra, lane, 0), Bt[0], acc);
    acc = WMMA_F16(load_a_tile(ra, lane, 1), Bt[1], acc);
    acc = WMMA_F16(load_a_tile(rb, lane, 0), Bt[2], acc);
    acc = WMMA_F16(load_a_tile(rb, lane, 1), Bt[3], acc);
    if (nl < Kk) {
#pragma unroll
      for (int r = 0; r < 8; ++r) {
        const int row = tile * 16 + r + 8 * g;
        pot[(size_t)row * Kk + nl] = acc[r] + bias;
      }
    }
  }
}

// ---------------------------------------------------------------------------
// Viterbi decode: one lane per batch row. K=9 -> 81 VALU MACs per step,
// sequential over T. trans cached in LDS; backpointers packed 9x4 bits into
// one u64 per (t,b) -> single global_store_b64 per step.
// ---------------------------------------------------------------------------
__global__ __launch_bounds__(256) void viterbi_kernel(
    const int* __restrict__ tokens, const float* __restrict__ pot,
    const float* __restrict__ trans, unsigned long long* __restrict__ bpws,
    float* __restrict__ dec, float* __restrict__ slen) {
  __shared__ float tr[Kk * Kk];
  const int tid = threadIdx.x;
  if (tid < Kk * Kk) tr[tid] = trans[tid];
  __syncthreads();

  const int b = blockIdx.x * 256 + tid;
  if (b >= Bb) return;

  const float* pb = pot + (size_t)b * Tt * Kk;
  float alpha[Kk];
#pragma unroll
  for (int k = 0; k < Kk; ++k) alpha[k] = pb[k];
  int cnt = (tokens[b * Tt] != 0) ? 1 : 0;

  for (int t = 1; t < Tt; ++t) {
    const bool m = (tokens[b * Tt + t] != 0);
    cnt += m ? 1 : 0;
    float na[Kk];
    unsigned long long bpword = 0ull;
#pragma unroll
    for (int k = 0; k < Kk; ++k) {
      float best = alpha[0] + tr[0 * Kk + k];
      int bj = 0;
#pragma unroll
      for (int j = 1; j < Kk; ++j) {
        const float s = alpha[j] + tr[j * Kk + k];
        if (s > best) { best = s; bj = j; }
      }
      const int sel = m ? bj : k;
      na[k] = m ? (best + pb[(size_t)t * Kk + k]) : alpha[k];
      bpword |= (unsigned long long)sel << (4 * k);
    }
#pragma unroll
    for (int k = 0; k < Kk; ++k) alpha[k] = na[k];
    bpws[(size_t)t * Bb + b] = bpword;
  }

  int last = 0;
  float bestA = alpha[0];
#pragma unroll
  for (int k = 1; k < Kk; ++k)
    if (alpha[k] > bestA) { bestA = alpha[k]; last = k; }

  int tag = last;
  dec[(size_t)b * Tt + (Tt - 1)] = (tokens[b * Tt + Tt - 1] != 0) ? (float)tag : 0.f;
  for (int t = Tt - 2; t >= 0; --t) {
    const unsigned long long w = bpws[(size_t)(t + 1) * Bb + b];
    tag = (int)((w >> (4 * tag)) & 0xFull);
    dec[(size_t)b * Tt + t] = (tokens[b * Tt + t] != 0) ? (float)tag : 0.f;
  }
  slen[b] = (float)cnt;
}

// ---------------------------------------------------------------------------
extern "C" void kernel_launch(void* const* d_in, const int* in_sizes, int n_in,
                              void* d_out, int out_size, void* d_ws, size_t ws_size,
                              hipStream_t stream) {
  const int*   tokens = (const int*)d_in[0];
  const float* emb    = (const float*)d_in[1];
  const float* Wf     = (const float*)d_in[2];
  const float* Uf     = (const float*)d_in[3];
  const float* bf     = (const float*)d_in[4];
  const float* Wb     = (const float*)d_in[5];
  const float* Ub     = (const float*)d_in[6];
  const float* bb     = (const float*)d_in[7];
  const float* Wd     = (const float*)d_in[8];
  const float* bd     = (const float*)d_in[9];
  const float* trans  = (const float*)d_in[10];

  float* outF = (float*)d_out;
  float* dec  = outF;                                   // [B,T]
  float* pot  = outF + (size_t)Bb * Tt;                 // [B,T,K]
  float* slen = pot + (size_t)Bb * Tt * Kk;             // [B]

  // Workspace: hf/hb f16 [B*T, 64], then packed backpointers u64 [T, B].
  _Float16* hf = (_Float16*)d_ws;
  _Float16* hb = hf + (size_t)Bb * Tt * 64;
  unsigned long long* bpws = (unsigned long long*)(hb + (size_t)Bb * Tt * 64);

  bilstm_kernel<<<64, 256, 0, stream>>>(tokens, emb, Wf, Uf, bf, Wb, Ub, bb, hf, hb);
  potentials_kernel<<<64, 256, 0, stream>>>(hf, hb, Wd, bd, pot);
  viterbi_kernel<<<2, 256, 0, stream>>>(tokens, pot, trans, bpws, dec, slen);
}
